// _Regression_head_77481210020385
// MI455X (gfx1250) — matrix operand order repaired and checked by reading, not compiled, not run on hardware
//
#include <hip/hip_runtime.h>
#include <hip/hip_bf16.h>

typedef _Float16 half_t;
typedef __attribute__((ext_vector_type(16))) _Float16 v16h;
typedef __attribute__((ext_vector_type(8)))  _Float16 v8h;
typedef __attribute__((ext_vector_type(8)))  float    v8f;

#define WMMA_F16(a, b, c) \
  __builtin_amdgcn_wmma_f32_16x16x32_f16(false, (a), false, (b), (short)0, (c), false, false)

// Problem constants
#define BATCH 64
#define CIN   2304
#define HW    196
#define DIM   512
#define MTOT  12544      // BATCH * HW
#define K3    4608       // DIM * 9

// ---------------------------------------------------------------------------
// CDNA5 async global->LDS copy (ASYNCcnt-tracked, no VGPR landing zone).
__device__ __forceinline__ void async_copy_b128(unsigned lds_byte_addr,
                                                const half_t* gptr) {
  asm volatile("global_load_async_to_lds_b128 %0, %1, off"
               :: "v"(lds_byte_addr), "v"((unsigned long long)gptr)
               : "memory");
}

template <int N>
__device__ __forceinline__ void wait_asynccnt() {
  asm volatile("s_wait_asynccnt %0" :: "i"(N) : "memory");
}

__device__ __forceinline__ unsigned lds_addr_of(const half_t* p) {
  // Low 32 bits of a flat pointer into LDS are the LDS byte offset.
  return (unsigned)(unsigned long long)p;
}

// ---------------------------------------------------------------------------
// fp32 -> f16 flat convert (w_reduce)
__global__ __launch_bounds__(256) void cvt_f32_f16(const float* __restrict__ in,
                                                   half_t* __restrict__ out, int n) {
  int i = blockIdx.x * 256 + threadIdx.x;
  if (i < n) out[i] = (half_t)in[i];
}

// 3x3 weight: w[d][c][j] (OIHW, j=ky*3+kx) -> Bt[d][j*512 + c] f16
__global__ __launch_bounds__(256) void cvt_w3(const float* __restrict__ in,
                                              half_t* __restrict__ out) {
  int d = blockIdx.x, j = blockIdx.y;
  for (int c = threadIdx.x; c < DIM; c += 256)
    out[(size_t)d * K3 + j * DIM + c] = (half_t)in[((size_t)d * DIM + c) * 9 + j];
}

// x (B, Cin, H, W) f32 -> A1[m = b*196+hw][c] f16 (transpose c<->hw per batch)
__global__ __launch_bounds__(256) void pack_x(const float* __restrict__ x,
                                              half_t* __restrict__ A1) {
  int m = blockIdx.x;
  int c = blockIdx.y * 256 + threadIdx.x;   // grid.y = 9 -> 2304
  int b = m / HW, hw = m % HW;
  A1[(size_t)m * CIN + c] = (half_t)x[((size_t)b * CIN + c) * HW + hw];
}

// ---------------------------------------------------------------------------
// Generic f16 WMMA GEMM: C[M,N] = A[M,K] * Bt[N,K]^T, f32 accumulate.
// Double-buffered LDS fed by async global->LDS b128 copies; K-step 64.
// MODE 0: store f32 C row-major.
// MODE 1: store f16 to C16[(hw*64 + b)*512 + n]   (per-pixel Q/K panels)
// MODE 2: store f16 to C16[(hw*512 + n)*64 + b]   (per-pixel V^T panels)
template <int MODE>
__global__ __launch_bounds__(256) void gemm16(const half_t* __restrict__ A,
                                              const half_t* __restrict__ Bt,
                                              float* __restrict__ C,
                                              half_t* __restrict__ C16,
                                              int M, int N, int K) {
  // stride 72 f16 = 144B = 9*16B: keeps b128 alignment, rotates bank phase
  __shared__ half_t Als[2][128 * 72];   // 128 rows x 64 k per buffer
  __shared__ half_t Bls[2][64 * 72];    // 64 cols x 64 k per buffer
  const int t = threadIdx.x;
  const int m0 = blockIdx.x * 128;
  const int n0 = blockIdx.y * 64;
  const int wv = t >> 5, lane = t & 31;
  const int mw = (wv & 3) * 32, nw = (wv >> 2) * 32;
  const int lr = lane & 15, lh = lane >> 4;

  v8f acc[2][2] = {};

  // Issue one K-step-64 tile's async copies: 4 A chunks + 2 B chunks / thread.
  auto stage = [&](int buf, int kt) {
#pragma unroll
    for (int i = 0; i < 4; ++i) {
      int q = t + i * 256;               // 1024 chunks: 128 rows x 8 segs
      int row = q >> 3, seg = q & 7;
      async_copy_b128(lds_addr_of(&Als[buf][row * 72 + seg * 8]),
                      &A[(size_t)(m0 + row) * K + kt + seg * 8]);
    }
#pragma unroll
    for (int i = 0; i < 2; ++i) {
      int q = t + i * 256;               // 512 chunks: 64 rows x 8 segs
      int row = q >> 3, seg = q & 7;
      async_copy_b128(lds_addr_of(&Bls[buf][row * 72 + seg * 8]),
                      &Bt[(size_t)(n0 + row) * K + kt + seg * 8]);
    }
  };

  stage(0, 0);
  const int T = K >> 6;                  // K/64 (K is a multiple of 64)
  for (int it = 0; it < T; ++it) {
    const int buf = it & 1;
    if (it + 1 < T) {
      stage(buf ^ 1, (it + 1) * 64);     // prefetch next tile
      wait_asynccnt<6>();                // 6 newest = next tile; current tile done
    } else {
      wait_asynccnt<0>();
    }
    __syncthreads();

#pragma unroll
    for (int s = 0; s < 2; ++s) {        // two 32-wide K sub-steps
#pragma unroll
      for (int mi = 0; mi < 2; ++mi) {
        int r = mw + mi * 16 + lr;
        v16h a;
        *(v8h*)&a       = *(const v8h*)&Als[buf][r * 72 + s * 32 + lh * 8];
        *((v8h*)&a + 1) = *(const v8h*)&Als[buf][r * 72 + s * 32 + 16 + lh * 8];
#pragma unroll
        for (int ni = 0; ni < 2; ++ni) {
          int c = nw + ni * 16 + lr;
          v16h b;
          *(v8h*)&b       = *(const v8h*)&Bls[buf][c * 72 + s * 32 + lh * 8];
          *((v8h*)&b + 1) = *(const v8h*)&Bls[buf][c * 72 + s * 32 + 16 + lh * 8];
          acc[mi][ni] = WMMA_F16(a, b, acc[mi][ni]);
        }
      }
    }
    __syncthreads();                     // protect buf before it is re-staged
  }

#pragma unroll
  for (int mi = 0; mi < 2; ++mi)
#pragma unroll
    for (int ni = 0; ni < 2; ++ni)
#pragma unroll
      for (int e = 0; e < 8; ++e) {
        int m = m0 + mw + mi * 16 + lh * 8 + e;
        int n = n0 + nw + ni * 16 + lr;
        float val = acc[mi][ni][e];
        if (MODE == 0) {
          C[(size_t)m * N + n] = val;
        } else {
          int b = m / HW, hw = m % HW;
          if (MODE == 1)
            C16[((size_t)hw * BATCH + b) * DIM + n] = (half_t)val;
          else
            C16[((size_t)hw * DIM + n) * BATCH + b] = (half_t)val;
        }
      }
}

// ---------------------------------------------------------------------------
// im2col for 3x3 SAME conv: src f32 [m][512] -> dst f16 [m][j*512+c].
// GN=true fuses GroupNorm(1 group) + ReLU using per-sample stats.
template <bool GN>
__global__ __launch_bounds__(256) void im2col3x3(const float* __restrict__ src,
                                                 half_t* __restrict__ dst,
                                                 const float* __restrict__ stats,
                                                 const float* __restrict__ gamma,
                                                 const float* __restrict__ beta) {
  int m = blockIdx.x;     // 0..12543
  int j = blockIdx.y;     // 0..8
  int b = m / HW, hw = m % HW;
  int h = hw / 14, w = hw % 14;
  int hh = h + j / 3 - 1, ww = w + j % 3 - 1;
  bool oob = ((unsigned)hh >= 14u) || ((unsigned)ww >= 14u);
  const float* in = src + ((size_t)b * HW + hh * 14 + ww) * DIM;
  half_t* out = dst + (size_t)m * K3 + j * DIM;
  float mean = 0.f, rstd = 1.f;
  if (GN) { mean = stats[b * 2]; rstd = stats[b * 2 + 1]; }
  for (int c = threadIdx.x; c < DIM; c += 256) {
    float v = 0.f;
    if (!oob) {
      v = in[c];
      if (GN) {
        v = (v - mean) * rstd * gamma[c] + beta[c];
        v = fmaxf(v, 0.f);
      }
    }
    out[c] = (half_t)v;
  }
}

// ---------------------------------------------------------------------------
// Per-pixel cross-batch attention. One block per pixel (196 blocks, 8 waves).
// S = Q K^T / sqrt(512) -> softmax over j -> virt = S @ V.
__global__ __launch_bounds__(256) void attention_kernel(const half_t* __restrict__ Qp,
                                                        const half_t* __restrict__ Kp,
                                                        const half_t* __restrict__ Vt,
                                                        float* __restrict__ virt) {
  __shared__ float  S[64 * 68];     // scores, padded stride
  __shared__ half_t Att[64 * 72];   // softmaxed f16, stride 72 (144B = 9*16B)
  const int hw = blockIdx.x;
  const int t = threadIdx.x, wv = t >> 5, lane = t & 31;
  const int lr = lane & 15, lh = lane >> 4;
  const half_t* Q  = Qp + (size_t)hw * BATCH * DIM;
  const half_t* Km = Kp + (size_t)hw * BATCH * DIM;

  // stage 1: S = Q K^T * (1/sqrt(512)); wave wv does m-tile wv&3, n-tiles {2*(wv>>2), +1}
  {
    int mt = wv & 3, nt0 = (wv >> 2) * 2;
    v8f acc[2] = {};
    for (int kk = 0; kk < DIM; kk += 32) {
      v16h a;
      int r = mt * 16 + lr;
      *(v8h*)&a       = *(const v8h*)&Q[r * DIM + kk + lh * 8];
      *((v8h*)&a + 1) = *(const v8h*)&Q[r * DIM + kk + 16 + lh * 8];
#pragma unroll
      for (int ni = 0; ni < 2; ++ni) {
        int c = (nt0 + ni) * 16 + lr;
        v16h b;
        *(v8h*)&b       = *(const v8h*)&Km[c * DIM + kk + lh * 8];
        *((v8h*)&b + 1) = *(const v8h*)&Km[c * DIM + kk + 16 + lh * 8];
        acc[ni] = WMMA_F16(a, b, acc[ni]);
      }
    }
    const float scale = 0.04419417382415922f;  // 1/sqrt(512)
#pragma unroll
    for (int ni = 0; ni < 2; ++ni)
#pragma unroll
      for (int e = 0; e < 8; ++e) {
        int r = mt * 16 + lh * 8 + e, c = (nt0 + ni) * 16 + lr;
        S[r * 68 + c] = acc[ni][e] * scale;
      }
  }
  __syncthreads();

  // softmax over j (64 values per row), one thread per row
  if (t < 64) {
    float mx = -1e30f;
    for (int j = 0; j < 64; ++j) mx = fmaxf(mx, S[t * 68 + j]);
    float sum = 0.f;
    for (int j = 0; j < 64; ++j) {
      float e = __expf(S[t * 68 + j] - mx);
      S[t * 68 + j] = e;
      sum += e;
    }
    float inv = 1.f / sum;
    for (int j = 0; j < 64; ++j) Att[t * 72 + j] = (half_t)(S[t * 68 + j] * inv);
  }
  __syncthreads();

  // stage 2: virt = Att(64x64) @ V(64x512); K = 64 (two WMMA steps)
  {
    int mt = wv & 3;
    int r = mt * 16 + lr;
    v16h a0, a1;
    *(v8h*)&a0       = *(const v8h*)&Att[r * 72 + lh * 8];
    *((v8h*)&a0 + 1) = *(const v8h*)&Att[r * 72 + 16 + lh * 8];
    *(v8h*)&a1       = *(const v8h*)&Att[r * 72 + 32 + lh * 8];
    *((v8h*)&a1 + 1) = *(const v8h*)&Att[r * 72 + 48 + lh * 8];
    for (int it = 0; it < 16; ++it) {
      int nt = (wv >> 2) + 2 * it;         // 32 n-tiles split across wave groups
      int c = nt * 16 + lr;
      const half_t* Vc = Vt + ((size_t)hw * DIM + c) * BATCH;
      v8f acc = {};
      v16h b;
      *(v8h*)&b       = *(const v8h*)&Vc[lh * 8];
      *((v8h*)&b + 1) = *(const v8h*)&Vc[16 + lh * 8];
      acc = WMMA_F16(a0, b, acc);
      *(v8h*)&b       = *(const v8h*)&Vc[32 + lh * 8];
      *((v8h*)&b + 1) = *(const v8h*)&Vc[48 + lh * 8];
      acc = WMMA_F16(a1, b, acc);
#pragma unroll
      for (int e = 0; e < 8; ++e) {
        int i = mt * 16 + lh * 8 + e;      // batch index
        virt[((size_t)i * HW + hw) * DIM + nt * 16 + lr] = acc[e];
      }
    }
  }
}

// ---------------------------------------------------------------------------
// Per-sample mean / rstd over (C,H,W) for GroupNorm(1 group)
__global__ __launch_bounds__(256) void gn_stats(const float* __restrict__ virt,
                                                float* __restrict__ stats) {
  int b = blockIdx.x;
  const float* p = virt + (size_t)b * HW * DIM;
  float s = 0.f, s2 = 0.f;
  for (int i = threadIdx.x; i < HW * DIM; i += 256) {
    float v = p[i];
    s += v; s2 += v * v;
  }
  __shared__ float rs[256], rs2[256];
  rs[threadIdx.x] = s; rs2[threadIdx.x] = s2;
  __syncthreads();
  for (int off = 128; off > 0; off >>= 1) {
    if (threadIdx.x < off) {
      rs[threadIdx.x]  += rs[threadIdx.x + off];
      rs2[threadIdx.x] += rs2[threadIdx.x + off];
    }
    __syncthreads();
  }
  if (threadIdx.x == 0) {
    const float inv_n = 1.f / (float)(HW * DIM);
    float mean = rs[0] * inv_n;
    float var = rs2[0] * inv_n - mean * mean;
    stats[b * 2] = mean;
    stats[b * 2 + 1] = rsqrtf(var + 1e-5f);
  }
}

// residual add + global avg pool + linear head -> out[b]
__global__ __launch_bounds__(256) void final_head(const float* __restrict__ xr,
                                                  const float* __restrict__ co,
                                                  const float* __restrict__ wpred,
                                                  float* __restrict__ out) {
  int b = blockIdx.x;
  size_t base = (size_t)b * HW * DIM;
  float acc = 0.f;
  for (int i = threadIdx.x; i < HW * DIM; i += 256) {
    int c = i & (DIM - 1);
    acc += (xr[base + i] + co[base + i]) * wpred[c];
  }
  __shared__ float rs[256];
  rs[threadIdx.x] = acc;
  __syncthreads();
  for (int off = 128; off > 0; off >>= 1) {
    if (threadIdx.x < off) rs[threadIdx.x] += rs[threadIdx.x + off];
    __syncthreads();
  }
  if (threadIdx.x == 0) out[b] = rs[0] * (1.f / (float)HW);
}

// ---------------------------------------------------------------------------
extern "C" void kernel_launch(void* const* d_in, const int* in_sizes, int n_in,
                              void* d_out, int out_size, void* d_ws, size_t ws_size,
                              hipStream_t stream) {
  (void)in_sizes; (void)n_in; (void)out_size; (void)ws_size;
  const float* x        = (const float*)d_in[0];
  const float* w_reduce = (const float*)d_in[1];
  const float* w_q      = (const float*)d_in[2];
  const float* w_k      = (const float*)d_in[3];
  const float* w_v      = (const float*)d_in[4];
  const float* w_conv   = (const float*)d_in[5];
  const float* gamma    = (const float*)d_in[6];
  const float* beta     = (const float*)d_in[7];
  const float* w_pred   = (const float*)d_in[8];
  float* out = (float*)d_out;

  char* ws = (char*)d_ws;
  size_t off = 0;
  auto alloc = [&](size_t bytes) -> char* {
    char* p = ws + off;
    off += (bytes + 255) & ~(size_t)255;
    return p;
  };

  half_t* Abig = (half_t*)alloc((size_t)MTOT * K3 * 2);       // im2col buffer (reused)
  half_t* A1   = (half_t*)alloc((size_t)MTOT * CIN * 2);      // reduce-GEMM A; later Q/K/V panels
  half_t* Qp = A1;                                            // [196][64][512] f16
  half_t* Kp = A1 + (size_t)HW * BATCH * DIM;                 // same size
  half_t* Vt = A1 + (size_t)2 * HW * BATCH * DIM;             // [196][512][64] f16
  half_t* wr16 = (half_t*)alloc((size_t)DIM * CIN * 2);
  half_t* wq16 = (half_t*)alloc((size_t)DIM * K3 * 2);
  half_t* wk16 = (half_t*)alloc((size_t)DIM * K3 * 2);
  half_t* wv16 = (half_t*)alloc((size_t)DIM * K3 * 2);
  half_t* wc16 = (half_t*)alloc((size_t)DIM * K3 * 2);
  float* x_red = (float*)alloc((size_t)MTOT * DIM * 4);
  float* virt  = (float*)alloc((size_t)MTOT * DIM * 4);
  float* conv_out = virt;  // virt is dead once im2col<GN> consumed it
  float* stats = (float*)alloc(512);

  // weight conversions
  cvt_f32_f16<<<dim3((DIM * CIN + 255) / 256), 256, 0, stream>>>(w_reduce, wr16, DIM * CIN);
  cvt_w3<<<dim3(DIM, 9), 256, 0, stream>>>(w_q, wq16);
  cvt_w3<<<dim3(DIM, 9), 256, 0, stream>>>(w_k, wk16);
  cvt_w3<<<dim3(DIM, 9), 256, 0, stream>>>(w_v, wv16);
  cvt_w3<<<dim3(DIM, 9), 256, 0, stream>>>(w_conv, wc16);

  // 1x1 reduce conv as GEMM
  pack_x<<<dim3(MTOT, CIN / 256), 256, 0, stream>>>(x, A1);
  gemm16<0><<<dim3(MTOT / 128, DIM / 64), 256, 0, stream>>>(A1, wr16, x_red, (half_t*)nullptr,
                                                            MTOT, DIM, CIN);

  // q, k, v 3x3 convs via shared im2col
  im2col3x3<false><<<dim3(MTOT, 9), 256, 0, stream>>>(x_red, Abig, nullptr, nullptr, nullptr);
  gemm16<1><<<dim3(MTOT / 128, DIM / 64), 256, 0, stream>>>(Abig, wq16, (float*)nullptr, Qp,
                                                            MTOT, DIM, K3);
  gemm16<1><<<dim3(MTOT / 128, DIM / 64), 256, 0, stream>>>(Abig, wk16, (float*)nullptr, Kp,
                                                            MTOT, DIM, K3);
  gemm16<2><<<dim3(MTOT / 128, DIM / 64), 256, 0, stream>>>(Abig, wv16, (float*)nullptr, Vt,
                                                            MTOT, DIM, K3);

  // per-pixel cross-batch attention
  attention_kernel<<<dim3(HW), 256, 0, stream>>>(Qp, Kp, Vt, virt);

  // GroupNorm stats, then fused GN+ReLU into im2col for final conv
  gn_stats<<<dim3(BATCH), 256, 0, stream>>>(virt, stats);
  im2col3x3<true><<<dim3(MTOT, 9), 256, 0, stream>>>(virt, Abig, stats, gamma, beta);
  gemm16<0><<<dim3(MTOT / 128, DIM / 64), 256, 0, stream>>>(Abig, wc16, conv_out,
                                                            (half_t*)nullptr, MTOT, DIM, K3);

  // residual + pool + linear
  final_head<<<dim3(BATCH), 256, 0, stream>>>(x_red, conv_out, w_pred, out);
}
